// HybridGATModel_47906065219695
// MI455X (gfx1250) — matrix operand order repaired and checked by reading, not compiled, not run on hardware
//
#include <hip/hip_runtime.h>
#include <hip/hip_bf16.h>

typedef _Float16 f16;
typedef __attribute__((ext_vector_type(16))) _Float16 v16h;
typedef __attribute__((ext_vector_type(8)))  _Float16 v8h;
typedef __attribute__((ext_vector_type(8)))  float    v8f;

static __device__ inline v16h cat16(v8h a, v8h b) {
  return __builtin_shufflevector(a, b, 0,1,2,3,4,5,6,7,8,9,10,11,12,13,14,15);
}
static __device__ inline v8f wmma16x16x32(v16h a, v16h b, v8f c) {
  // D = A(16x32 f16) x B(32x16 f16) + C(f32)
  return __builtin_amdgcn_wmma_f32_16x16x32_f16(false, a, false, b, (short)0, c, false, false);
}

// ---------------------------------------------------------------------------
// Prep kernels (fold BN into conv weights, layout conversions, f32->f16)
// ---------------------------------------------------------------------------

// images NCHW f32 (16,3,128,128) -> NHWC f16 (16,128,128,32) zero-padded channels
__global__ void prep_images(const float* __restrict__ img, f16* __restrict__ out) {
  size_t idx = (size_t)blockIdx.x * blockDim.x + threadIdx.x;
  const size_t total = (size_t)16 * 128 * 128 * 32;
  if (idx >= total) return;
  int c = idx % 32; size_t t = idx / 32;
  int x = t % 128; t /= 128;
  int y = t % 128; int b = (int)(t / 128);
  float v = 0.f;
  if (c < 3) v = img[(((size_t)b * 3 + c) * 128 + y) * 128 + x];
  out[idx] = (f16)v;
}

// OIHW f32 conv weights + BN(eval) -> folded f16 [9][Cout][CinPad] + f32 bias[Cout]
__global__ void prep_convw(const float* __restrict__ w, const float* __restrict__ cb,
                           const float* __restrict__ g, const float* __restrict__ bb,
                           const float* __restrict__ mm, const float* __restrict__ vv,
                           f16* __restrict__ wt, float* __restrict__ biasf,
                           int Cout, int CinR, int CinP) {
  size_t idx = (size_t)blockIdx.x * blockDim.x + threadIdx.x;
  const size_t total = (size_t)9 * Cout * CinP;
  if (idx >= total) return;
  int ci = idx % CinP; size_t t = idx / CinP;
  int co = t % Cout; int tap = (int)(t / Cout);
  float s = g[co] * rsqrtf(vv[co] + 1e-5f);
  float val = 0.f;
  if (ci < CinR) val = w[((size_t)co * CinR + ci) * 9 + tap] * s;
  wt[idx] = (f16)val;
  if (idx < (size_t)Cout) {
    float s2 = g[idx] * rsqrtf(vv[idx] + 1e-5f);
    biasf[idx] = cb[idx] * s2 + bb[idx] - mm[idx] * s2;
  }
}

// GAT weight f32 [K][M] -> f16 [M][K] (transposed, row-K contiguous for fragments)
__global__ void prep_gw(const float* __restrict__ gw, f16* __restrict__ wt, int K, int M) {
  size_t idx = (size_t)blockIdx.x * blockDim.x + threadIdx.x;
  if (idx >= (size_t)K * M) return;
  int m2 = idx % M; int k = (int)(idx / M);
  wt[(size_t)m2 * K + k] = (f16)gw[idx];
}

// ---------------------------------------------------------------------------
// Implicit-GEMM 3x3 SAME conv + bias + ReLU, NHWC f16, WMMA f32 accumulate.
// 8-wave workgroup: 128 pixels x 64 Cout. Weight chunk (64x32 f16, 4KB) is
// staged into double-buffered LDS with GLOBAL_LOAD_ASYNC_TO_LDS_B128 (ASYNCcnt)
// and overlapped with WMMA compute; A-fragments then come from LDS (ds_load).
// ---------------------------------------------------------------------------
__global__ __launch_bounds__(256) void conv3x3_bn_relu_wmma(
    const f16* __restrict__ in,   // [B][H][W][Cin]
    const f16* __restrict__ wt,   // [9][Cout][Cin]  (BN-folded)
    const float* __restrict__ bias,
    f16* __restrict__ out,        // [B][H][W][Cout]
    int H, int W, int Cin, int Cout)
{
  __shared__ f16 smem[2][64 * 32];          // 2 x 4KB weight tiles
  const int tid  = threadIdx.x;
  const int lane = tid & 31;
  const int wave = tid >> 5;
  const int m  = lane & 15;                 // A row in tile / B,D column
  const int hh = lane >> 4;                 // lane half -> K sub-ranges (ISA layout)
  const int p  = (blockIdx.x * 8 + wave) * 16 + m;  // this lane's pixel
  const int HW = H * W;
  const int b  = p / HW;
  const int rem = p - b * HW;
  const int y = rem / W;
  const int x = rem - y * W;
  const int co0 = blockIdx.y * 64;

  const int kc = Cin >> 5;                  // 32-wide K chunks per tap
  const int NC = 9 * kc;

  // Per-thread async staging: 16B each; 256 threads cover 64 rows x 64B.
  const int sr = tid >> 2, sseg = tid & 3;
  const f16* wbase = wt + (size_t)(co0 + sr) * Cin + sseg * 8;
  auto stage = [&](int tap, int k0, int bufIdx) {
    const f16* gp = wbase + (size_t)tap * Cout * Cin + k0;
    unsigned ldsOff = (unsigned)(unsigned long long)&smem[bufIdx][sr * 32 + sseg * 8];
    asm volatile("global_load_async_to_lds_b128 %0, %1, off"
                 :: "v"(ldsOff), "v"((unsigned long long)gp)
                 : "memory");
  };

  v8f acc[4] = {};
  stage(0, 0, 0);
  int tap = 0, k0 = 0;
  for (int c = 0; c < NC; ++c) {
    asm volatile("s_wait_asynccnt 0x0" ::: "memory");  // my async done
    __syncthreads();                                   // => everyone's async done

    // next chunk indices (incremental: no division in the loop)
    int ntap = tap, nk0 = k0 + 32;
    if (nk0 == Cin) { nk0 = 0; ntap = tap + 1; }
    if (c + 1 < NC) stage(ntap, nk0, (c + 1) & 1);     // overlap next DMA

    const int dy = tap / 3 - 1, dx = tap % 3 - 1;
    const int yy = y + dy, xx = x + dx;
    const bool inb = ((unsigned)yy < (unsigned)H) && ((unsigned)xx < (unsigned)W);
    v16h bf;
    if (inb) {
      const f16* bp = in + ((size_t)(b * H + yy) * W + xx) * Cin + k0 + hh * 8;
      bf = cat16(*(const v8h*)bp, *(const v8h*)(bp + 16));
    } else {
      v16h z = {};
      bf = z;                                          // SAME zero padding
    }
    // load all 4 A fragments first (clustered ds_load_b128, one dscnt wait),
    // then issue the 4 WMMAs back-to-back
    v16h af[4];
#pragma unroll
    for (int t = 0; t < 4; ++t) {
      const f16* arow = &smem[c & 1][(t * 16 + m) * 32];
      af[t] = cat16(*(const v8h*)(arow + hh * 8),
                    *(const v8h*)(arow + 16 + hh * 8));
    }
#pragma unroll
    for (int t = 0; t < 4; ++t) acc[t] = wmma16x16x32(af[t], bf, acc[t]);

    tap = ntap; k0 = nk0;
  }

  f16* orow = out + (size_t)p * Cout;
#pragma unroll
  for (int t = 0; t < 4; ++t) {
    const int co = co0 + t * 16 + hh * 8;              // D: m = hh*8 + r
    v8h vals;
#pragma unroll
    for (int r = 0; r < 8; ++r) {
      float v = acc[t][r] + bias[co + r];
      vals[r] = (f16)(v > 0.f ? v : 0.f);
    }
    *(v8h*)(orow + co) = vals;                         // contiguous b128 store
  }
}

// ---------------------------------------------------------------------------
// Generic WMMA GEMM: out[n][m] = sum_k A[m][k] * X[n][k]  (+bias[m], ReLU)
// A:[M][K], X:[N][K] row-major f16; out:[N][M] f16. Wave: 16 N x 64 M.
// ---------------------------------------------------------------------------
__global__ __launch_bounds__(32) void gemm_nt_wmma(
    const f16* __restrict__ Am, const f16* __restrict__ Xm,
    const float* __restrict__ bias, f16* __restrict__ out,
    int M, int K, int doRelu)
{
  const int lane = threadIdx.x;
  const int m  = lane & 15;
  const int hh = lane >> 4;
  const int n  = blockIdx.x * 16 + m;
  const int m0 = blockIdx.y * 64;

  v8f acc[4] = {};
  const f16* xrow = Xm + (size_t)n * K;
  for (int k0 = 0; k0 < K; k0 += 32) {
    const f16* bp = xrow + k0 + hh * 8;
    v16h bf = cat16(*(const v8h*)bp, *(const v8h*)(bp + 16));
    v16h af[4];
#pragma unroll
    for (int t = 0; t < 4; ++t) {
      const f16* arow = Am + (size_t)(m0 + t * 16 + m) * K + k0;
      af[t] = cat16(*(const v8h*)(arow + hh * 8),
                    *(const v8h*)(arow + 16 + hh * 8));
    }
#pragma unroll
    for (int t = 0; t < 4; ++t) acc[t] = wmma16x16x32(af[t], bf, acc[t]);
  }
  f16* orow = out + (size_t)n * M;
#pragma unroll
  for (int t = 0; t < 4; ++t) {
    const int c = m0 + t * 16 + hh * 8;
    v8h vals;
#pragma unroll
    for (int r = 0; r < 8; ++r) {
      float v = acc[t][r] + (bias ? bias[c + r] : 0.f);
      if (doRelu) v = v > 0.f ? v : 0.f;
      vals[r] = (f16)v;
    }
    *(v8h*)(orow + c) = vals;
  }
}

// ---------------------------------------------------------------------------
// 2x2 maxpool NHWC f16
// ---------------------------------------------------------------------------
__global__ void maxpool2x2(const f16* __restrict__ in, f16* __restrict__ out,
                           int H, int W, int C, size_t total) {
  size_t idx = (size_t)blockIdx.x * blockDim.x + threadIdx.x;
  if (idx >= total) return;
  int c = idx % C; size_t t = idx / C;
  int x = t % (W / 2); t /= (W / 2);
  int y = t % (H / 2); int b = (int)(t / (H / 2));
  const f16* base = in + (((size_t)(b * H + 2 * y) * W) + 2 * x) * C + c;
  float v0 = (float)base[0], v1 = (float)base[C];
  float v2 = (float)base[(size_t)W * C], v3 = (float)base[(size_t)W * C + C];
  out[idx] = (f16)fmaxf(fmaxf(v0, v1), fmaxf(v2, v3));
}

// ---------------------------------------------------------------------------
// GAT pieces
// ---------------------------------------------------------------------------
__global__ __launch_bounds__(32) void att_logits(
    const f16* __restrict__ h, const float* __restrict__ as_,
    const float* __restrict__ ad_, float* __restrict__ als,
    float* __restrict__ ald, int C) {
  int node = blockIdx.x;
  const f16* row = h + (size_t)node * C;
  float s0 = 0.f, s1 = 0.f;
  for (int c = threadIdx.x; c < C; c += 32) {
    float hv = (float)row[c];
    s0 += hv * as_[c];
    s1 += hv * ad_[c];
  }
  for (int off = 16; off > 0; off >>= 1) {
    s0 += __shfl_down(s0, off, 32);
    s1 += __shfl_down(s1, off, 32);
  }
  if (threadIdx.x == 0) { als[node] = s0; ald[node] = s1; }
}

// P[d][s] = softmax_s( leaky_relu(al_s[s] + al_d[d]) )  for d,s in 0..255
__global__ __launch_bounds__(256) void att_softmax(
    const float* __restrict__ als, const float* __restrict__ ald,
    f16* __restrict__ P) {
  __shared__ float red[256];
  const int d = blockIdx.x, s = threadIdx.x;
  float e = als[s] + ald[d];
  e = e < 0.f ? 0.2f * e : e;
  red[s] = e; __syncthreads();
  for (int off = 128; off > 0; off >>= 1) {
    if (s < off) red[s] = fmaxf(red[s], red[s + off]);
    __syncthreads();
  }
  float mx = red[0]; __syncthreads();
  float pr = __expf(e - mx);
  red[s] = pr; __syncthreads();
  for (int off = 128; off > 0; off >>= 1) {
    if (s < off) red[s] += red[s + off];
    __syncthreads();
  }
  P[(size_t)d * 256 + s] = (f16)(pr / red[0]);
}

// naive f16 transpose: in [R][C] -> out [C][R]
__global__ void transpose_f16(const f16* __restrict__ in, f16* __restrict__ out,
                              int R, int C) {
  size_t idx = (size_t)blockIdx.x * blockDim.x + threadIdx.x;
  if (idx >= (size_t)R * C) return;
  int c = idx % C; int r = (int)(idx / C);
  out[(size_t)c * R + r] = in[idx];
}

// self-loop-only nodes (256..4095): out = relu(h + bias)
__global__ void gat_selfloops(const f16* __restrict__ h, const float* __restrict__ bias,
                              f16* __restrict__ out, int C) {
  size_t idx = (size_t)blockIdx.x * blockDim.x + threadIdx.x;
  const size_t total = (size_t)(4096 - 256) * C;
  if (idx >= total) return;
  int c = idx % C; size_t d = 256 + idx / C;
  float v = (float)h[d * C + c] + bias[c];
  out[d * C + c] = (f16)(v > 0.f ? v : 0.f);
}

// per-image mean over 256 nodes: in [16][256][128] -> out [16][128] f32
__global__ __launch_bounds__(128) void mean_pool(const f16* __restrict__ in,
                                                 float* __restrict__ out) {
  int b = blockIdx.x, c = threadIdx.x;
  float s = 0.f;
  for (int n = 0; n < 256; ++n) s += (float)in[((size_t)b * 256 + n) * 128 + c];
  out[b * 128 + c] = s * (1.f / 256.f);
}

// final 128->10 linear + log_softmax
__global__ __launch_bounds__(32) void head_logsoftmax(
    const float* __restrict__ pooled, const float* __restrict__ ow,
    const float* __restrict__ ob, float* __restrict__ out) {
  int b = blockIdx.x;
  if (threadIdx.x != 0) return;
  float lg[10]; float mx = -1e30f;
  for (int c = 0; c < 10; ++c) {
    float s = ob[c];
    for (int k = 0; k < 128; ++k) s += pooled[b * 128 + k] * ow[k * 10 + c];
    lg[c] = s; mx = fmaxf(mx, s);
  }
  float se = 0.f;
  for (int c = 0; c < 10; ++c) se += __expf(lg[c] - mx);
  float lse = mx + __logf(se);
  for (int c = 0; c < 10; ++c) out[b * 10 + c] = lg[c] - lse;
}

// ---------------------------------------------------------------------------
extern "C" void kernel_launch(void* const* d_in, const int* in_sizes, int n_in,
                              void* d_out, int out_size, void* d_ws, size_t ws_size,
                              hipStream_t stream) {
  (void)in_sizes; (void)n_in; (void)out_size;
  const float* images = (const float*)d_in[0];
  const float* cw[4], *cb[4], *bg[4], *bb[4], *bm[4], *bv[4];
  for (int l = 0; l < 4; ++l) {
    int base = 1 + l * 6;
    cw[l] = (const float*)d_in[base + 0];
    cb[l] = (const float*)d_in[base + 1];
    bg[l] = (const float*)d_in[base + 2];
    bb[l] = (const float*)d_in[base + 3];
    bm[l] = (const float*)d_in[base + 4];
    bv[l] = (const float*)d_in[base + 5];
  }
  const float* g1_w  = (const float*)d_in[25];
  const float* g1_as = (const float*)d_in[26];
  const float* g1_ad = (const float*)d_in[27];
  const float* g1_b  = (const float*)d_in[28];
  const float* g2_w  = (const float*)d_in[29];
  const float* g2_as = (const float*)d_in[30];
  const float* g2_ad = (const float*)d_in[31];
  const float* g2_b  = (const float*)d_in[32];
  const float* out_w = (const float*)d_in[33];
  const float* out_b = (const float*)d_in[34];

  // ---- workspace arena (regions reused across the conv pipeline) ----
  char* ws = (char*)d_ws;
  size_t off = 0;
  auto alloc = [&](size_t bytes) -> char* {
    char* p = ws + off;
    off = (off + bytes + 255) & ~(size_t)255;
    return p;
  };
  char* regA = alloc((size_t)16 * 128 * 128 * 32 * 2);   // imgP -> act2p -> act4
  char* regB = alloc((size_t)16 * 128 * 128 * 64 * 2);   // act1 -> act3 -> act4p
  char* regC = alloc((size_t)16 * 128 * 128 * 128 * 2);  // act2 -> act3p
  f16* wt1 = (f16*)alloc((size_t)9 * 64 * 32 * 2);
  f16* wt2 = (f16*)alloc((size_t)9 * 128 * 64 * 2);
  f16* wt3 = (f16*)alloc((size_t)9 * 256 * 128 * 2);
  f16* wt4 = (f16*)alloc((size_t)9 * 256 * 256 * 2);
  float* bias1 = (float*)alloc(64 * 4);
  float* bias2 = (float*)alloc(128 * 4);
  float* bias3 = (float*)alloc(256 * 4);
  float* bias4 = (float*)alloc(256 * 4);
  f16* gw1 = (f16*)alloc((size_t)256 * 256 * 2);
  f16* gw2 = (f16*)alloc((size_t)128 * 256 * 2);
  f16* h1    = (f16*)alloc((size_t)4096 * 256 * 2);
  f16* hT1   = (f16*)alloc((size_t)256 * 256 * 2);
  f16* P1    = (f16*)alloc((size_t)256 * 256 * 2);
  f16* gout1 = (f16*)alloc((size_t)4096 * 256 * 2);
  f16* h2    = (f16*)alloc((size_t)4096 * 128 * 2);
  f16* hT2   = (f16*)alloc((size_t)128 * 256 * 2);
  f16* P2    = (f16*)alloc((size_t)256 * 256 * 2);
  f16* gout2 = (f16*)alloc((size_t)4096 * 128 * 2);
  float* als1 = (float*)alloc(256 * 4);
  float* ald1 = (float*)alloc(256 * 4);
  float* als2 = (float*)alloc(256 * 4);
  float* ald2 = (float*)alloc(256 * 4);
  float* pooled = (float*)alloc((size_t)16 * 128 * 4);
  if (off > ws_size) return;  // workspace too small: bail deterministically

  f16* imgP  = (f16*)regA;
  f16* act1  = (f16*)regB;
  f16* act2  = (f16*)regC;
  f16* act2p = (f16*)regA;
  f16* act3  = (f16*)regB;
  f16* act3p = (f16*)regC;
  f16* act4  = (f16*)regA;
  f16* act4p = (f16*)regB;

  auto g1 = [](size_t total) { return dim3((unsigned)((total + 255) / 256)); };

  // ---- prep ----
  prep_images<<<g1((size_t)16 * 128 * 128 * 32), 256, 0, stream>>>(images, imgP);
  prep_convw<<<g1((size_t)9 * 64 * 32), 256, 0, stream>>>(cw[0], cb[0], bg[0], bb[0], bm[0], bv[0], wt1, bias1, 64, 3, 32);
  prep_convw<<<g1((size_t)9 * 128 * 64), 256, 0, stream>>>(cw[1], cb[1], bg[1], bb[1], bm[1], bv[1], wt2, bias2, 128, 64, 64);
  prep_convw<<<g1((size_t)9 * 256 * 128), 256, 0, stream>>>(cw[2], cb[2], bg[2], bb[2], bm[2], bv[2], wt3, bias3, 256, 128, 128);
  prep_convw<<<g1((size_t)9 * 256 * 256), 256, 0, stream>>>(cw[3], cb[3], bg[3], bb[3], bm[3], bv[3], wt4, bias4, 256, 256, 256);
  prep_gw<<<g1((size_t)256 * 256), 256, 0, stream>>>(g1_w, gw1, 256, 256);
  prep_gw<<<g1((size_t)256 * 128), 256, 0, stream>>>(g2_w, gw2, 256, 128);

  // ---- CNN (WMMA implicit GEMM, async-LDS weight staging) ----
  conv3x3_bn_relu_wmma<<<dim3(16 * 128 * 128 / 128, 1), 256, 0, stream>>>(imgP, wt1, bias1, act1, 128, 128, 32, 64);
  conv3x3_bn_relu_wmma<<<dim3(16 * 128 * 128 / 128, 2), 256, 0, stream>>>(act1, wt2, bias2, act2, 128, 128, 64, 128);
  maxpool2x2<<<g1((size_t)16 * 64 * 64 * 128), 256, 0, stream>>>(act2, act2p, 128, 128, 128, (size_t)16 * 64 * 64 * 128);
  conv3x3_bn_relu_wmma<<<dim3(16 * 64 * 64 / 128, 4), 256, 0, stream>>>(act2p, wt3, bias3, act3, 64, 64, 128, 256);
  maxpool2x2<<<g1((size_t)16 * 32 * 32 * 256), 256, 0, stream>>>(act3, act3p, 64, 64, 256, (size_t)16 * 32 * 32 * 256);
  conv3x3_bn_relu_wmma<<<dim3(16 * 32 * 32 / 128, 4), 256, 0, stream>>>(act3p, wt4, bias4, act4, 32, 32, 256, 256);
  maxpool2x2<<<g1((size_t)16 * 16 * 16 * 256), 256, 0, stream>>>(act4, act4p, 32, 32, 256, (size_t)16 * 16 * 16 * 256);

  // ---- GAT layer 1 (act4p == node features [4096][256], NHWC==node-major) ----
  gemm_nt_wmma<<<dim3(4096 / 16, 256 / 64), 32, 0, stream>>>(gw1, act4p, nullptr, h1, 256, 256, 0);
  att_logits<<<256, 32, 0, stream>>>(h1, g1_as, g1_ad, als1, ald1, 256);
  att_softmax<<<256, 256, 0, stream>>>(als1, ald1, P1);
  transpose_f16<<<g1((size_t)256 * 256), 256, 0, stream>>>(h1, hT1, 256, 256);
  // rows 0..255: out = P1 @ h1[0:256] + b, relu   (A = hT1 [c][s], X = P1 [d][s])
  gemm_nt_wmma<<<dim3(256 / 16, 256 / 64), 32, 0, stream>>>(hT1, P1, g1_b, gout1, 256, 256, 1);
  gat_selfloops<<<g1((size_t)(4096 - 256) * 256), 256, 0, stream>>>(h1, g1_b, gout1, 256);

  // ---- GAT layer 2 ----
  gemm_nt_wmma<<<dim3(4096 / 16, 128 / 64), 32, 0, stream>>>(gw2, gout1, nullptr, h2, 128, 256, 0);
  att_logits<<<256, 32, 0, stream>>>(h2, g2_as, g2_ad, als2, ald2, 128);
  att_softmax<<<256, 256, 0, stream>>>(als2, ald2, P2);
  transpose_f16<<<g1((size_t)256 * 128), 256, 0, stream>>>(h2, hT2, 256, 128);
  gemm_nt_wmma<<<dim3(256 / 16, 128 / 64), 32, 0, stream>>>(hT2, P2, g2_b, gout2, 128, 256, 1);
  gat_selfloops<<<g1((size_t)(4096 - 256) * 128), 256, 0, stream>>>(h2, g2_b, gout2, 128);

  // ---- pool + head ----
  mean_pool<<<16, 128, 0, stream>>>(gout2, pooled);
  head_logsoftmax<<<16, 32, 0, stream>>>(pooled, out_w, out_b, (float*)d_out);
}
